// PhotoVerseAttnProcessor_13134009991635
// MI455X (gfx1250) — compile-verified, matrix operand order checked by reading, not run on hardware
//
#include <hip/hip_runtime.h>
#include <stdint.h>

#define B_    8
#define S_    4096
#define HID_  1280
#define HEADS_ 20
#define DH_   64
#define TXT_  77
#define IPN_  5
#define CROSS_ 768

typedef __attribute__((ext_vector_type(16))) __bf16   v16bf;
typedef __attribute__((ext_vector_type(8)))  float    v8f;
typedef __attribute__((ext_vector_type(8)))  uint32_t v8u;

static __device__ __forceinline__ uint16_t f32_to_bf16(float f) {
  uint32_t u = __float_as_uint(f);
  uint32_t r = (u + 0x7FFFu + ((u >> 16) & 1u)) >> 16;
  return (uint16_t)r;
}

static __device__ __forceinline__ v8f zero_acc() {
  v8f z;
  for (int i = 0; i < 8; ++i) z[i] = 0.0f;
  return z;
}

static __device__ __forceinline__ v16bf zero_frag() {
  v8u z;
  for (int i = 0; i < 8; ++i) z[i] = 0u;
  return __builtin_bit_cast(v16bf, z);
}

// Load a 16x32 bf16 A/B fragment for one lane. `base` points at element
// (row_of_this_lane, kbase_of_this_lane). ISA layout: VGPR j holds K-pairs
// {0,1,2,3} then {8,9,10,11} relative to the lane's K base -> two b128 loads.
static __device__ __forceinline__ v16bf load_frag(const uint16_t* base) {
  const uint4* p = reinterpret_cast<const uint4*>(base);
  uint4 a = p[0];   // pairs 0..3  (K = kb+0..7)
  uint4 b = p[2];   // pairs 8..11 (K = kb+16..23)
  v8u r;
  r[0] = a.x; r[1] = a.y; r[2] = a.z; r[3] = a.w;
  r[4] = b.x; r[5] = b.y; r[6] = b.z; r[7] = b.w;
  return __builtin_bit_cast(v16bf, r);
}

static __device__ __forceinline__ v8f wmma_bf16(v16bf a, v16bf b, v8f c) {
  return __builtin_amdgcn_wmma_f32_16x16x32_bf16(false, a, false, b, (short)0, c,
                                                 false, false);
}

// ---------------------------------------------------------------- conversions
__global__ void cvt_f32_to_bf16(const float* __restrict__ in,
                                uint16_t* __restrict__ out, size_t n) {
  size_t i = (size_t)blockIdx.x * blockDim.x + threadIdx.x;
  size_t stride = (size_t)gridDim.x * blockDim.x;
  for (; i < n; i += stride) out[i] = f32_to_bf16(in[i]);
}

// out[n*K + k] = bf16(in[k*N + n])  (transpose so GEMM B-operand K-pairs are contiguous)
__global__ void tcvt_f32_to_bf16(const float* __restrict__ in,
                                 uint16_t* __restrict__ out, int K, int N) {
  size_t total = (size_t)K * N;
  size_t i = (size_t)blockIdx.x * blockDim.x + threadIdx.x;
  size_t stride = (size_t)gridDim.x * blockDim.x;
  for (; i < total; i += stride) {
    int k = (int)(i % K);
    int n = (int)(i / K);
    out[i] = f32_to_bf16(in[(size_t)k * N + n]);
  }
}

// ------------------------------------------------------------------- GEMM
// C[M,N] = A[M,K] (bf16) @ Wt[N,K]^T (bf16), acc f32.
// Block = 256 threads = 8 waves (4 in M x 2 in N); wave tile = 32x64.
// Block tile = 128x128. N must be a multiple of 128 (true here: N=1280).
__global__ __launch_bounds__(256) void gemm_bf16_wmma(
    const uint16_t* __restrict__ A, const uint16_t* __restrict__ Wt,
    uint16_t* __restrict__ Cb, float* __restrict__ Cf,
    const float* __restrict__ bias, int M, int N, int K, int f32out) {
  const int lane = threadIdx.x & 31;
  const int wave = threadIdx.x >> 5;
  const int wm = wave & 3, wn = wave >> 2;
  const int m0 = blockIdx.y * 128 + wm * 32;
  const int n0 = blockIdx.x * 128 + wn * 64;
  const int lm = lane & 15;
  const int koff = (lane < 16) ? 0 : 8;

  v8f acc[2][4];
  for (int mt = 0; mt < 2; ++mt)
    for (int nt = 0; nt < 4; ++nt) acc[mt][nt] = zero_acc();

  size_t arow[2], brow[4];
  for (int mt = 0; mt < 2; ++mt) {
    int r = m0 + mt * 16 + lm;
    if (r >= M) r = M - 1;            // clamp (stores are guarded)
    arow[mt] = (size_t)r * K;
  }
  for (int nt = 0; nt < 4; ++nt) {
    int c = n0 + nt * 16 + lm;
    if (c >= N) c = N - 1;
    brow[nt] = (size_t)c * K;
  }

#pragma unroll 2
  for (int k0 = 0; k0 < K; k0 += 32) {
    __builtin_prefetch(A + arow[0] + k0 + 256, 0, 1);
    v16bf af[2], bf[4];
    for (int mt = 0; mt < 2; ++mt) af[mt] = load_frag(A + arow[mt] + k0 + koff);
    for (int nt = 0; nt < 4; ++nt) bf[nt] = load_frag(Wt + brow[nt] + k0 + koff);
    for (int mt = 0; mt < 2; ++mt)
      for (int nt = 0; nt < 4; ++nt)
        acc[mt][nt] = wmma_bf16(af[mt], bf[nt], acc[mt][nt]);
  }

  const int mrow = (lane >> 4) * 8;   // C layout: VGPR v -> M = v (+8 for hi lanes)
  for (int mt = 0; mt < 2; ++mt)
    for (int nt = 0; nt < 4; ++nt) {
      int n = n0 + nt * 16 + lm;
      if (n >= N) continue;
      for (int v = 0; v < 8; ++v) {
        int m = m0 + mt * 16 + v + mrow;
        if (m >= M) continue;
        size_t idx = (size_t)m * N + n;
        float val = acc[mt][nt][v];
        if (f32out) Cf[idx] = val + bias[n];
        else        Cb[idx] = f32_to_bf16(val);
      }
    }
}

// --------------------------------------------------------------- attention
// Per wave: one 16-query tile of one (b, head). 77 text keys + 5 IP keys.
// All 8 waves of a block share (b, head): V tiles staged transposed in LDS.
__global__ __launch_bounds__(256) void attn_fused(
    const uint16_t* __restrict__ Q,    // [B,S,HID] bf16
    const uint16_t* __restrict__ Kt,   // [B,TXT,HID]
    const uint16_t* __restrict__ Vt,   // [B,TXT,HID]
    const uint16_t* __restrict__ Ki,   // [B,IPN,HID]
    const uint16_t* __restrict__ Vi,   // [B,IPN,HID]
    uint16_t* __restrict__ Out) {      // [B,S,HID] bf16
  __shared__ __align__(16) uint16_t probs[8][16][96];     // padded to 96 keys
  __shared__ __align__(16) uint16_t probs_ip[8][16][32];  // padded to 32 keys
  __shared__ __align__(16) uint16_t vt_lds[DH_][96];      // V^T  [d][key], zero-padded
  __shared__ __align__(16) uint16_t vi_lds[DH_][32];      // Vip^T[d][key], zero-padded

  const int lane = threadIdx.x & 31;
  const int wave = threadIdx.x >> 5;
  const int qt = blockIdx.x * 8 + wave;
  const int h  = blockIdx.y;
  const int b  = blockIdx.z;
  const int lm = lane & 15;
  const int hi = lane >> 4;
  const int koff = hi ? 8 : 0;
  const float scale = 0.125f;   // DH^-0.5

  // ---- cooperative stage of V^T tiles (coalesced reads, one-time transpose)
  for (int i = threadIdx.x; i < 96 * DH_; i += 256) {
    int key = i >> 6, d = i & 63;
    vt_lds[d][key] = (key < TXT_)
        ? Vt[((size_t)b * TXT_ + key) * HID_ + h * DH_ + d] : (uint16_t)0;
  }
  for (int i = threadIdx.x; i < 32 * DH_; i += 256) {
    int key = i >> 6, d = i & 63;
    vi_lds[d][key] = (key < IPN_)
        ? Vi[((size_t)b * IPN_ + key) * HID_ + h * DH_ + d] : (uint16_t)0;
  }
  __syncthreads();

  // Q A-fragments: rows = queries, K = head dim (2 steps of 32)
  const size_t qbase = ((size_t)b * S_ + (size_t)qt * 16 + lm) * HID_ + h * DH_;
  v16bf qf[2];
  qf[0] = load_frag(Q + qbase + koff);
  qf[1] = load_frag(Q + qbase + 32 + koff);

  // ---- text scores: 5 key tiles of 16
  v8f sc[5];
  for (int t = 0; t < 5; ++t) {
    sc[t] = zero_acc();
    int key = t * 16 + lm;
    v16bf kf0 = zero_frag(), kf1 = zero_frag();
    if (key < TXT_) {
      size_t kb = ((size_t)b * TXT_ + key) * HID_ + h * DH_;
      kf0 = load_frag(Kt + kb + koff);
      kf1 = load_frag(Kt + kb + 32 + koff);
    }
    sc[t] = wmma_bf16(qf[0], kf0, sc[t]);
    sc[t] = wmma_bf16(qf[1], kf1, sc[t]);
  }

  // ---- IP scores: 1 key tile
  v8f si = zero_acc();
  {
    v16bf kf0 = zero_frag(), kf1 = zero_frag();
    if (lm < IPN_) {
      size_t kb = ((size_t)b * IPN_ + lm) * HID_ + h * DH_;
      kf0 = load_frag(Ki + kb + koff);
      kf1 = load_frag(Ki + kb + 32 + koff);
    }
    si = wmma_bf16(qf[0], kf0, si);
    si = wmma_bf16(qf[1], kf1, si);
  }

  // ---- softmax (C layout: N across 16 lanes, M = v + 8*hi)
  for (int v = 0; v < 8; ++v) {
    int m = v + hi * 8;
    // text branch
    float mx = -3.0e38f;
    for (int t = 0; t < 5; ++t) {
      int key = t * 16 + lm;
      float s = (key < TXT_) ? sc[t][v] * scale : -3.0e38f;
      sc[t][v] = s;
      mx = fmaxf(mx, s);
    }
    for (int o = 8; o; o >>= 1) mx = fmaxf(mx, __shfl_xor(mx, o, 16));
    float sum = 0.0f;
    for (int t = 0; t < 5; ++t) {
      float p = __expf(sc[t][v] - mx);
      sc[t][v] = p;
      sum += p;
    }
    for (int o = 8; o; o >>= 1) sum += __shfl_xor(sum, o, 16);
    float inv = 1.0f / sum;
    for (int t = 0; t < 5; ++t)
      probs[wave][m][t * 16 + lm] = f32_to_bf16(sc[t][v] * inv);
    probs[wave][m][80 + lm] = 0;   // pad keys 80..95

    // IP branch
    float s = (lm < IPN_) ? si[v] * scale : -3.0e38f;
    float mxi = s;
    for (int o = 8; o; o >>= 1) mxi = fmaxf(mxi, __shfl_xor(mxi, o, 16));
    float p = __expf(s - mxi);
    float sumi = p;
    for (int o = 8; o; o >>= 1) sumi += __shfl_xor(sumi, o, 16);
    probs_ip[wave][m][lm] = f32_to_bf16(p / sumi);
    probs_ip[wave][m][16 + lm] = 0;  // pad keys 16..31
  }

  // ---- P @ V for both branches, combine txt + 3*ip, store bf16
  for (int dt = 0; dt < 4; ++dt) {
    int d = dt * 16 + lm;               // B/C column = output dim
    v8f ot = zero_acc(), oi = zero_acc();
    for (int ks = 0; ks < 3; ++ks) {    // 96 padded text keys
      v16bf pf = load_frag(&probs[wave][lm][ks * 32 + koff]);
      v16bf vf = load_frag(&vt_lds[d][ks * 32 + koff]);
      ot = wmma_bf16(pf, vf, ot);
    }
    {
      v16bf pf = load_frag(&probs_ip[wave][lm][koff]);
      v16bf vf = load_frag(&vi_lds[d][koff]);
      oi = wmma_bf16(pf, vf, oi);
    }
    for (int v = 0; v < 8; ++v) {
      int m = v + hi * 8;
      size_t row = (size_t)b * S_ + (size_t)qt * 16 + m;
      Out[row * HID_ + h * DH_ + d] = f32_to_bf16(ot[v] + 3.0f * oi[v]);
    }
  }
}

// ---------------------------------------------------------------- launch
extern "C" void kernel_launch(void* const* d_in, const int* in_sizes, int n_in,
                              void* d_out, int out_size, void* d_ws, size_t ws_size,
                              hipStream_t stream) {
  const float* hs   = (const float*)d_in[0];
  const float* enc  = (const float*)d_in[1];
  const float* ip   = (const float*)d_in[2];
  const float* Wq   = (const float*)d_in[3];
  const float* Wk   = (const float*)d_in[4];
  const float* Wv   = (const float*)d_in[5];
  const float* Wkip = (const float*)d_in[6];
  const float* Wvip = (const float*)d_in[7];
  const float* Wo   = (const float*)d_in[8];
  const float* bo   = (const float*)d_in[9];
  float* out = (float*)d_out;

  uint8_t* base = (uint8_t*)d_ws;
  size_t off = 0;
  auto alloc = [&](size_t elems) -> uint16_t* {
    uint16_t* p = (uint16_t*)(base + off);
    off += (elems * 2 + 255) & ~(size_t)255;
    return p;
  };
  uint16_t* hsb   = alloc((size_t)B_ * S_ * HID_);  // later reused as attn out
  uint16_t* Qb    = alloc((size_t)B_ * S_ * HID_);
  uint16_t* encb  = alloc((size_t)B_ * TXT_ * CROSS_);
  uint16_t* ipb   = alloc((size_t)B_ * IPN_ * CROSS_);
  uint16_t* WqT   = alloc((size_t)HID_ * HID_);
  uint16_t* WoT   = alloc((size_t)HID_ * HID_);
  uint16_t* WkT   = alloc((size_t)CROSS_ * HID_);
  uint16_t* WvT   = alloc((size_t)CROSS_ * HID_);
  uint16_t* WkiT  = alloc((size_t)CROSS_ * HID_);
  uint16_t* WviT  = alloc((size_t)CROSS_ * HID_);
  uint16_t* Ktxt  = alloc((size_t)B_ * TXT_ * HID_);
  uint16_t* Vtxt  = alloc((size_t)B_ * TXT_ * HID_);
  uint16_t* Kip   = alloc((size_t)B_ * IPN_ * HID_);
  uint16_t* Vip   = alloc((size_t)B_ * IPN_ * HID_);

  hipLaunchKernelGGL(cvt_f32_to_bf16, dim3(4096), dim3(256), 0, stream,
                     hs, hsb, (size_t)B_ * S_ * HID_);
  hipLaunchKernelGGL(cvt_f32_to_bf16, dim3(512), dim3(256), 0, stream,
                     enc, encb, (size_t)B_ * TXT_ * CROSS_);
  hipLaunchKernelGGL(cvt_f32_to_bf16, dim3(64), dim3(256), 0, stream,
                     ip, ipb, (size_t)B_ * IPN_ * CROSS_);
  hipLaunchKernelGGL(tcvt_f32_to_bf16, dim3(2048), dim3(256), 0, stream, Wq, WqT, HID_, HID_);
  hipLaunchKernelGGL(tcvt_f32_to_bf16, dim3(2048), dim3(256), 0, stream, Wo, WoT, HID_, HID_);
  hipLaunchKernelGGL(tcvt_f32_to_bf16, dim3(2048), dim3(256), 0, stream, Wk, WkT, CROSS_, HID_);
  hipLaunchKernelGGL(tcvt_f32_to_bf16, dim3(2048), dim3(256), 0, stream, Wv, WvT, CROSS_, HID_);
  hipLaunchKernelGGL(tcvt_f32_to_bf16, dim3(2048), dim3(256), 0, stream, Wkip, WkiT, CROSS_, HID_);
  hipLaunchKernelGGL(tcvt_f32_to_bf16, dim3(2048), dim3(256), 0, stream, Wvip, WviT, CROSS_, HID_);

  dim3 blk(256);
  // Q = hs @ Wq
  hipLaunchKernelGGL(gemm_bf16_wmma, dim3(HID_ / 128, (B_ * S_) / 128), blk, 0, stream,
                     hsb, WqT, Qb, (float*)nullptr, (const float*)nullptr,
                     B_ * S_, HID_, HID_, 0);
  // K/V projections
  hipLaunchKernelGGL(gemm_bf16_wmma, dim3(HID_ / 128, (B_ * TXT_ + 127) / 128), blk, 0, stream,
                     encb, WkT, Ktxt, (float*)nullptr, (const float*)nullptr,
                     B_ * TXT_, HID_, CROSS_, 0);
  hipLaunchKernelGGL(gemm_bf16_wmma, dim3(HID_ / 128, (B_ * TXT_ + 127) / 128), blk, 0, stream,
                     encb, WvT, Vtxt, (float*)nullptr, (const float*)nullptr,
                     B_ * TXT_, HID_, CROSS_, 0);
  hipLaunchKernelGGL(gemm_bf16_wmma, dim3(HID_ / 128, 1), blk, 0, stream,
                     ipb, WkiT, Kip, (float*)nullptr, (const float*)nullptr,
                     B_ * IPN_, HID_, CROSS_, 0);
  hipLaunchKernelGGL(gemm_bf16_wmma, dim3(HID_ / 128, 1), blk, 0, stream,
                     ipb, WviT, Vip, (float*)nullptr, (const float*)nullptr,
                     B_ * IPN_, HID_, CROSS_, 0);
  // fused dual attention (writes into hsb, which is dead after the Q GEMM)
  hipLaunchKernelGGL(attn_fused, dim3(S_ / 16 / 8, HEADS_, B_), blk, 0, stream,
                     Qb, Ktxt, Vtxt, Kip, Vip, hsb);
  // out = attn @ Wo + bo  (f32 output)
  hipLaunchKernelGGL(gemm_bf16_wmma, dim3(HID_ / 128, (B_ * S_) / 128), blk, 0, stream,
                     hsb, WoT, (uint16_t*)nullptr, out, bo,
                     B_ * S_, HID_, HID_, 1);
}
